// ODEModel_14250701488512
// MI455X (gfx1250) — matrix operand order repaired and checked by reading, not compiled, not run on hardware
//
#include <hip/hip_runtime.h>

#define T_GRID 4096
#define HID 32

typedef __attribute__((ext_vector_type(16))) __bf16 v16bf;
typedef __attribute__((ext_vector_type(8)))  float  v8f;

union BfPack { v16bf v; unsigned u[8]; };

// float -> bf16 bits in low 16 (round-to-nearest-even)
#if __has_builtin(__builtin_amdgcn_cvt_pk_bf16_f32)
typedef __attribute__((ext_vector_type(2))) __bf16 v2bf;
__device__ __forceinline__ unsigned f2bf(float x) {
    union { v2bf v; unsigned u; } c;
    c.v = __builtin_amdgcn_cvt_pk_bf16_f32(x, 0.0f);   // single v_cvt_pk_bf16_f32
    return c.u & 0xffffu;
}
#else
__device__ __forceinline__ unsigned f2bf(float x) {
    unsigned u = __float_as_uint(x);
    return (u + 0x7fffu + ((u >> 16) & 1u)) >> 16;
}
#endif

__device__ __forceinline__ float fast_tanh(float x) {
#if __has_builtin(__builtin_amdgcn_tanhf)
    return __builtin_amdgcn_tanhf(x);          // single v_tanh_f32 (gfx1250 trans op)
#elif __has_builtin(__builtin_amdgcn_tanh_f32)
    return __builtin_amdgcn_tanh_f32(x);
#else
    float xc = fminf(fmaxf(x, -9.0f), 9.0f);
    float e  = __expf(2.0f * xc);              // v_exp_f32 path
    return (e - 1.0f) * __builtin_amdgcn_rcpf(e + 1.0f);
#endif
}

// sigmoid(x) = 0.5*tanh(x/2) + 0.5 : one transcendental instead of exp+rcp
__device__ __forceinline__ float fast_sigmoid(float x) {
    return fmaf(0.5f, fast_tanh(0.5f * x), 0.5f);
}

// wave-uniform broadcast from a literal lane: v_readlane (no DS round trip)
__device__ __forceinline__ float bcast(float v, int l) {
    return __int_as_float(__builtin_amdgcn_readlane(__float_as_int(v), l));
}

__global__ __launch_bounds__(32, 1)
void ode_rk4_wave(const float* __restrict__ s_grid,
                  const float* __restrict__ y0,
                  const float* __restrict__ W1, const float* __restrict__ b1,
                  const float* __restrict__ W2, const float* __restrict__ b2,
                  const float* __restrict__ W3, const float* __restrict__ b3,
                  float* __restrict__ out)
{
    const int lane = threadIdx.x;      // 0..31, wave32
    const int n    = lane & 15;        // column within 16-wide WMMA tile
    const int khw  = lane & 16;        // 16 if upper half-wave

    // 64-byte LDS staging buffers for the broadcast-A operand (bf16 h vector)
    __shared__ __align__(16) unsigned short hb1[32];
    __shared__ __align__(16) unsigned short hb2[32];

    // ---- layer-1 weights: lane j owns column j of W1 + bias ----
    float w1c0 = W1[0*HID + lane];
    float w1c1 = W1[1*HID + lane];
    float w1c2 = W1[2*HID + lane];
    float w1c3 = W1[3*HID + lane];
    float w1c4 = W1[4*HID + lane];
    float b1v  = b1[lane];

    // ---- W2 halves + padded W3 as bf16 B-operands (this code's contiguous k order, same as A:
    //      lane = column n, VGPR v = K pair {2v,2v+1} (+16 for upper half-wave)) ----
    BfPack B1, B2, B3;
#pragma unroll
    for (int v = 0; v < 8; ++v) {
        int k0 = 2*v + khw;
        B1.u[v] = f2bf(W2[k0*HID + n])      | (f2bf(W2[(k0+1)*HID + n])      << 16);
        B2.u[v] = f2bf(W2[k0*HID + n + 16]) | (f2bf(W2[(k0+1)*HID + n + 16]) << 16);
        float w3lo = (n < 4) ? W3[k0*4 + n]     : 0.0f;
        float w3hi = (n < 4) ? W3[(k0+1)*4 + n] : 0.0f;
        B3.u[v] = f2bf(w3lo) | (f2bf(w3hi) << 16);
    }

    // ---- biases folded into C operands (broadcast down each column) ----
    v8f c2a, c2b, c3;
    {
        float b2lo = b2[n], b2hi = b2[n + 16];
        float b3v  = (n < 4) ? b3[n] : 0.0f;
#pragma unroll
        for (int i = 0; i < 8; ++i) { c2a[i] = b2lo; c2b[i] = b2hi; c3[i] = b3v; }
    }

    // Build broadcast-A (every row = h^T, 16x32 bf16 layout) via LDS:
    // store h[lane] as bf16 at byte 2*lane, then two lane-constant b128 loads.
    // Low-half lanes read bytes [0..15] & [16..31]; high-half [32..47] & [48..63].
    // DS ops from one wave execute in order, so no barrier is needed.
    auto buildA = [&](float hval, unsigned short* buf) -> BfPack {
        buf[lane] = (unsigned short)f2bf(hval);
        const uint4* p = (const uint4*)buf;
        uint4 lo = p[khw >> 3];          // bytes 0..15  (or 32..47)
        uint4 hi = p[(khw >> 3) + 1];    // bytes 16..31 (or 48..63)
        BfPack A;
        A.u[0] = lo.x; A.u[1] = lo.y; A.u[2] = lo.z; A.u[3] = lo.w;
        A.u[4] = hi.x; A.u[5] = hi.y; A.u[6] = hi.z; A.u[7] = hi.w;
        return A;
    };

    // ---- wave-uniform RK state ----
    float ya0 = y0[0], ya1 = y0[1], ya2 = y0[2], ya3 = y0[3];
    if (lane < 4) {
        float yv = (lane == 0) ? ya0 : (lane == 1) ? ya1 : (lane == 2) ? ya2 : ya3;
        out[lane] = yv;   // row t=0
    }

    // f(y, s): returns k[j] in lane j (j = 0..3)
    auto mlp = [&](float z0, float z1, float z2, float z3, float s) -> float {
        // layer 1: per-lane column of W1 (z inputs are wave-uniform / SGPR)
        float a = fmaf(z0, w1c0, b1v);
        a = fmaf(z1, w1c1, a);
        a = fmaf(z2, w1c2, a);
        a = fmaf(z3, w1c3, a);
        a = fmaf(s,  w1c4, a);
        float h1 = fast_tanh(a);

        // h2 = tanh(W2^T h1 + b2): two independent WMMAs cover output cols 0-15 / 16-31
        BfPack A = buildA(h1, hb1);
        v8f d1 = __builtin_amdgcn_wmma_f32_16x16x32_bf16(false, A.v, false, B1.v,
                                                         (short)0, c2a, false, false);
        v8f d2 = __builtin_amdgcn_wmma_f32_16x16x32_bf16(false, A.v, false, B2.v,
                                                         (short)0, c2b, false, false);
        float h2 = fast_tanh((lane < 16) ? d1[0] : d2[0]);   // lane L holds h2[L]

        // layer 3: same broadcast-A trick, B = zero-padded W3
        BfPack A2 = buildA(h2, hb2);
        v8f d3 = __builtin_amdgcn_wmma_f32_16x16x32_bf16(false, A2.v, false, B3.v,
                                                         (short)0, c3, false, false);
        return fast_sigmoid(d3[0]);   // lane j<4 holds output component j
    };

#pragma clang loop unroll(disable)
    for (int t = 0; t < T_GRID - 1; ++t) {
        float s = s_grid[t];
        float h = s_grid[t + 1] - s;
        float hh = 0.5f * h;

        float k1 = mlp(ya0, ya1, ya2, ya3, s);
        float k10 = bcast(k1, 0), k11 = bcast(k1, 1), k12 = bcast(k1, 2), k13 = bcast(k1, 3);

        float k2 = mlp(fmaf(hh, k10, ya0), fmaf(hh, k11, ya1),
                       fmaf(hh, k12, ya2), fmaf(hh, k13, ya3), s);
        float k20 = bcast(k2, 0), k21 = bcast(k2, 1), k22 = bcast(k2, 2), k23 = bcast(k2, 3);

        float k3 = mlp(fmaf(hh, k20, ya0), fmaf(hh, k21, ya1),
                       fmaf(hh, k22, ya2), fmaf(hh, k23, ya3), s);
        float k30 = bcast(k3, 0), k31 = bcast(k3, 1), k32 = bcast(k3, 2), k33 = bcast(k3, 3);

        float k4 = mlp(fmaf(h, k30, ya0), fmaf(h, k31, ya1),
                       fmaf(h, k32, ya2), fmaf(h, k33, ya3), s);
        float k40 = bcast(k4, 0), k41 = bcast(k4, 1), k42 = bcast(k4, 2), k43 = bcast(k4, 3);

        float w = h * (1.0f / 6.0f);
        ya0 = fmaf(w, k10 + 2.0f*k20 + 2.0f*k30 + k40, ya0);
        ya1 = fmaf(w, k11 + 2.0f*k21 + 2.0f*k31 + k41, ya1);
        ya2 = fmaf(w, k12 + 2.0f*k22 + 2.0f*k32 + k42, ya2);
        ya3 = fmaf(w, k13 + 2.0f*k23 + 2.0f*k33 + k43, ya3);

        if (lane < 4) {
            float yv = (lane == 0) ? ya0 : (lane == 1) ? ya1 : (lane == 2) ? ya2 : ya3;
            out[(t + 1) * 4 + lane] = yv;
        }
    }
}

extern "C" void kernel_launch(void* const* d_in, const int* in_sizes, int n_in,
                              void* d_out, int out_size, void* d_ws, size_t ws_size,
                              hipStream_t stream) {
    (void)in_sizes; (void)n_in; (void)out_size; (void)d_ws; (void)ws_size;
    ode_rk4_wave<<<1, 32, 0, stream>>>(
        (const float*)d_in[0],   // s_grid [T]
        (const float*)d_in[1],   // y0 [4]
        (const float*)d_in[2],   // W1 [5,32]
        (const float*)d_in[3],   // b1 [32]
        (const float*)d_in[4],   // W2 [32,32]
        (const float*)d_in[5],   // b2 [32]
        (const float*)d_in[6],   // W3 [32,4]
        (const float*)d_in[7],   // b3 [4]
        (float*)d_out);          // out [T,4] f32
}
